// GroupQueryAttention_9844065042574
// MI455X (gfx1250) — matrix operand order repaired and checked by reading, not compiled, not run on hardware
//
#include <hip/hip_runtime.h>
#include <cstdint>

// ---------------------------------------------------------------------------
// GQA forward for MI455X (gfx1250): bf16 WMMA pipeline.
//   x[2,2048,2048] -> Q/K/V proj (WMMA gemm) -> flash attention (WMMA) ->
//   O proj (WMMA gemm, fp32 out + bias).
// Matmuls: v_wmma_f32_16x16x32_bf16. LDS staging uses CDNA5 async
// global->LDS copies (ASYNCcnt) with double buffering when available.
// ---------------------------------------------------------------------------

typedef __attribute__((ext_vector_type(16))) __bf16 bf16x16;
typedef __attribute__((ext_vector_type(8)))  float  f32x8;
typedef __attribute__((ext_vector_type(4)))  int    i32x4;

union FragBits { uint32_t u[8]; bf16x16 f; };

#if defined(__has_builtin)
#  if __has_builtin(__builtin_amdgcn_global_load_async_to_lds_b128) && \
      __has_builtin(__builtin_amdgcn_s_wait_asynccnt)
#    define HAVE_ASYNC_LDS 1
#  endif
#endif
#ifndef HAVE_ASYNC_LDS
#  define HAVE_ASYNC_LDS 0
#endif

#if HAVE_ASYNC_LDS
// Builtin signature (from hipcc diagnostic): param0 = int32x4 in AS1 (global),
// param1 = LDS side, then imm offset + imm cpol.
typedef __attribute__((address_space(1))) i32x4 as1_i32x4;
typedef __attribute__((address_space(3))) i32x4 as3_i32x4;
__device__ __forceinline__ void async_copy_b128(const uint32_t* g, uint32_t* l) {
  // GLOBAL_LOAD_ASYNC_TO_LDS_B128: per-lane 16B global->LDS, tracked by ASYNCcnt.
  __builtin_amdgcn_global_load_async_to_lds_b128((as1_i32x4*)g, (as3_i32x4*)l, 0, 0);
}
#  define ASYNC_WAIT() __builtin_amdgcn_s_wait_asynccnt(0)
#else
#  define ASYNC_WAIT() ((void)0)
#endif

// A-operand fragment: 16x32 bf16, rows (M) on lanes, K contiguous:
// lane-half hi: dword offsets {0..7} + 8*hi within a 16-dw K-chunk.
__device__ __forceinline__ bf16x16 gather_a(const uint32_t* base, int row,
                                            int stride_dw, int col_dw) {
  const int lane = threadIdx.x & 31;
  const int hi = lane >> 4;
  const uint32_t* p = base + (size_t)(row + (lane & 15)) * stride_dw + col_dw + (hi << 3);
  FragBits r;
#pragma unroll
  for (int v = 0; v < 8; ++v) r.u[v] = p[v];
  return r.f;
}

// B-operand fragment: 32x16 bf16, cols (N) on lanes, K contiguous:
// lane-half hi: dword offsets {0..7} + 8*hi (source stored as Bt[N][K]).
__device__ __forceinline__ bf16x16 gather_b(const uint32_t* base, int row,
                                            int stride_dw, int col_dw) {
  const int lane = threadIdx.x & 31;
  const int hi = lane >> 4;
  const uint32_t* p = base + (size_t)(row + (lane & 15)) * stride_dw + col_dw + (hi << 3);
  FragBits r;
#pragma unroll
  for (int v = 0; v < 8; ++v) r.u[v] = p[v];
  return r.f;
}

__device__ __forceinline__ f32x8 wmma_bf16(bf16x16 a, bf16x16 b, f32x8 c) {
  return __builtin_amdgcn_wmma_f32_16x16x32_bf16(false, a, false, b,
                                                 (short)0, c, false, false);
}

// ---------------------------------------------------------------------------
// fp32 -> bf16 elementwise convert (vectorized)
// ---------------------------------------------------------------------------
__global__ __launch_bounds__(256) void cvt_f32_bf16_kernel(
    const float* __restrict__ in, __bf16* __restrict__ out, size_t n) {
  size_t i = ((size_t)blockIdx.x * 256 + threadIdx.x) * 4;
  if (i + 3 < n) {
    float4 v = *(const float4*)(in + i);
    union { __bf16 h[4]; uint2 u; } pk;
    pk.h[0] = (__bf16)v.x; pk.h[1] = (__bf16)v.y;
    pk.h[2] = (__bf16)v.z; pk.h[3] = (__bf16)v.w;
    *(uint2*)(out + i) = pk.u;
  }
}

// ---------------------------------------------------------------------------
// W[K][N] fp32  ->  Wt[N][K] bf16  (LDS-tiled transpose + convert)
// ---------------------------------------------------------------------------
__global__ __launch_bounds__(256) void transpose_cvt_kernel(
    const float* __restrict__ in, __bf16* __restrict__ out, int Kdim, int Ndim) {
  __shared__ float t[32][33];
  const int tx = threadIdx.x & 31, ty = threadIdx.x >> 5;
  const int n0 = blockIdx.x * 32, k0 = blockIdx.y * 32;
#pragma unroll
  for (int i = ty; i < 32; i += 8)
    t[i][tx] = in[(size_t)(k0 + i) * Ndim + n0 + tx];
  __syncthreads();
#pragma unroll
  for (int i = ty; i < 32; i += 8)
    out[(size_t)(n0 + i) * Kdim + k0 + tx] = (__bf16)t[tx][i];
}

// ---------------------------------------------------------------------------
// V[b*s][512] bf16 -> Vt[(b*4+kv)*128 + d][2048] bf16 (per-head transpose)
// ---------------------------------------------------------------------------
__global__ __launch_bounds__(256) void transpose_v_kernel(
    const __bf16* __restrict__ V, __bf16* __restrict__ Vt) {
  __shared__ __bf16 t[32][34];
  const int tx = threadIdx.x & 31, ty = threadIdx.x >> 5;
  const int s0 = blockIdx.x * 32, d0 = blockIdx.y * 32;
  const int bkv = blockIdx.z, b = bkv >> 2, kvh = bkv & 3;
#pragma unroll
  for (int i = ty; i < 32; i += 8)
    t[i][tx] = V[(size_t)(b * 2048 + s0 + i) * 512 + kvh * 128 + d0 + tx];
  __syncthreads();
#pragma unroll
  for (int i = ty; i < 32; i += 8)
    Vt[(size_t)(bkv * 128 + d0 + i) * 2048 + s0 + tx] = t[tx][i];
}

// ---------------------------------------------------------------------------
// GEMM: C[M][N] = A[M][K](bf16) x Bt[N][K](bf16) + bias[N]
// block = 8 waves, block tile 128x128, wave tile 32x64 (2x4 WMMA -> 8 wmma
// per 6 fragment gathers), K step 32, double-buffered LDS, async staging.
// LDS row stride 20 dw: keeps b128 alignment, rows spread over 64 banks.
// ---------------------------------------------------------------------------
template <bool OUT_F32>
__global__ __launch_bounds__(256) void gemm_bias_kernel(
    const __bf16* __restrict__ A, const __bf16* __restrict__ Bt,
    const float* __restrict__ bias, void* __restrict__ Cout,
    int M, int N, int K) {
  __shared__ uint32_t ldsA[2][128 * 20];   // 128 rows x 16 dw, pad 4
  __shared__ uint32_t ldsB[2][128 * 20];
  const int tid = threadIdx.x;
  const int lane = tid & 31;
  const int w = tid >> 5;
  const int wm = (w & 3) * 32;
  const int wn = (w >> 2) * 64;
  const int gm0 = blockIdx.y * 128;
  const int gn0 = blockIdx.x * 128;
  const uint32_t* Ad = (const uint32_t*)A;
  const uint32_t* Bd = (const uint32_t*)Bt;
  const int Kd = K >> 1;

  f32x8 acc[8];
  f32x8 zero = {};
#pragma unroll
  for (int i = 0; i < 8; ++i) acc[i] = zero;

  const int r0 = tid >> 2;            // 0..63
  const int c4 = (tid & 3) << 2;      // 0,4,8,12

  auto stage = [&](uint32_t* dA, uint32_t* dB, int k0) {
    const int kdw = k0 >> 1;
    const uint32_t* sA0 = Ad + (size_t)(gm0 + r0) * Kd + kdw + c4;
    const uint32_t* sA1 = Ad + (size_t)(gm0 + r0 + 64) * Kd + kdw + c4;
    const uint32_t* sB0 = Bd + (size_t)(gn0 + r0) * Kd + kdw + c4;
    const uint32_t* sB1 = Bd + (size_t)(gn0 + r0 + 64) * Kd + kdw + c4;
#if HAVE_ASYNC_LDS
    async_copy_b128(sA0, &dA[r0 * 20 + c4]);
    async_copy_b128(sA1, &dA[(r0 + 64) * 20 + c4]);
    async_copy_b128(sB0, &dB[r0 * 20 + c4]);
    async_copy_b128(sB1, &dB[(r0 + 64) * 20 + c4]);
#else
    *(uint4*)&dA[r0 * 20 + c4]        = *(const uint4*)sA0;
    *(uint4*)&dA[(r0 + 64) * 20 + c4] = *(const uint4*)sA1;
    *(uint4*)&dB[r0 * 20 + c4]        = *(const uint4*)sB0;
    *(uint4*)&dB[(r0 + 64) * 20 + c4] = *(const uint4*)sB1;
#endif
  };

  const int nk = K >> 5;
  stage(ldsA[0], ldsB[0], 0);
  for (int kt = 0; kt < nk; ++kt) {
    ASYNC_WAIT();
    __syncthreads();   // single barrier per K-step: see ping-pong argument
    if (kt + 1 < nk) stage(ldsA[(kt + 1) & 1], ldsB[(kt + 1) & 1], (kt + 1) << 5);
    const uint32_t* bA = ldsA[kt & 1];
    const uint32_t* bB = ldsB[kt & 1];
    bf16x16 a0 = gather_a(bA, wm, 20, 0);
    bf16x16 a1 = gather_a(bA, wm + 16, 20, 0);
    bf16x16 b[4];
#pragma unroll
    for (int j = 0; j < 4; ++j) b[j] = gather_b(bB, wn + j * 16, 20, 0);
#pragma unroll
    for (int j = 0; j < 4; ++j) {
      acc[j]     = wmma_bf16(a0, b[j], acc[j]);
      acc[4 + j] = wmma_bf16(a1, b[j], acc[4 + j]);
    }
  }

  const int hi = lane >> 4, ln = lane & 15;
#pragma unroll
  for (int mi = 0; mi < 2; ++mi)
#pragma unroll
    for (int j = 0; j < 4; ++j)
#pragma unroll
      for (int r = 0; r < 8; ++r) {
        int m = gm0 + wm + mi * 16 + r + (hi << 3);
        int n = gn0 + wn + j * 16 + ln;
        float v = acc[mi * 4 + j][r] + bias[n];
        if constexpr (OUT_F32)
          ((float*)Cout)[(size_t)m * N + n] = v;
        else
          ((__bf16*)Cout)[(size_t)m * N + n] = (__bf16)v;
      }
}

// ---------------------------------------------------------------------------
// Flash attention: one block per (q-tile of 128, head, batch); 8 waves,
// 16 q-rows per wave; keys streamed in 64-wide chunks through double-buffered
// LDS (async); online softmax; P round-trips per-wave LDS scratch to reach
// A-fragment layout; V consumed pre-transposed as B-fragments from global.
// ---------------------------------------------------------------------------
__global__ __launch_bounds__(256) void flash_attn_kernel(
    const __bf16* __restrict__ Q,    // [b*s][2048]
    const __bf16* __restrict__ Kk,   // [b*s][512]
    const __bf16* __restrict__ Vt,   // [(b*4+kv)*128 + d][2048]
    __bf16* __restrict__ Oattn) {    // [b*s][2048]
  __shared__ uint32_t ldsK[2][64 * 68];    // 64 keys x 64 dw, pad 4 (b128 ok)
  __shared__ uint32_t ldsP[8 * 16 * 33];   // per-wave 16 q x 32 dw, pad 1
  const int tid = threadIdx.x;
  const int lane = tid & 31;
  const int w = tid >> 5;
  const int hi = lane >> 4, ln = lane & 15;
  const int qt = blockIdx.x;
  const int h  = blockIdx.y;
  const int bi = blockIdx.z;
  const int kv = h >> 2;                   // repeat_interleave(4)
  const int q0 = qt * 128 + w * 16;

  const uint32_t* Qd = (const uint32_t*)Q;
  const uint32_t* Kd = (const uint32_t*)Kk;
  const uint32_t* Vd = (const uint32_t*)Vt;

  // Preload 16x128 Q tile as 4 A-fragments (per-wave, held in VGPRs).
  bf16x16 qf[4];
  {
    const uint32_t* qbase = Qd + (size_t)(bi * 2048 + q0) * 1024 + (h * 128 >> 1);
#pragma unroll
    for (int c = 0; c < 4; ++c) qf[c] = gather_a(qbase, 0, 1024, c * 16);
  }

  f32x8 o[8];
  f32x8 zero = {};
#pragma unroll
  for (int i = 0; i < 8; ++i) o[i] = zero;
  float mrun[8], lrun[8];
#pragma unroll
  for (int r = 0; r < 8; ++r) { mrun[r] = -3.0e38f; lrun[r] = 0.0f; }

  const float scale = 0.08838834764831845f;  // 1/sqrt(128)
  uint32_t* pw = ldsP + w * (16 * 33);
  __bf16* pwh = (__bf16*)pw;

  auto stage_k = [&](uint32_t* dst, int kc2) {
#pragma unroll
    for (int t = 0; t < 4; ++t) {
      int i = tid + t * 256;
      int r = i >> 4, cc = (i & 15) << 2;
      const uint32_t* src =
          Kd + (size_t)(bi * 2048 + kc2 * 64 + r) * 256 + kv * 64 + cc;
#if HAVE_ASYNC_LDS
      async_copy_b128(src, &dst[r * 68 + cc]);
#else
      *(uint4*)&dst[r * 68 + cc] = *(const uint4*)src;
#endif
    }
  };

  stage_k(ldsK[0], 0);
  for (int kc = 0; kc < 32; ++kc) {          // 2048 keys / 64
    ASYNC_WAIT();
    __syncthreads();
    if (kc + 1 < 32) {
      stage_k(ldsK[(kc + 1) & 1], kc + 1);
      // Prefetch next V chunk (128 d-rows x 32 dw) into cache.
      const uint32_t* vnext = Vd + (size_t)((bi * 4 + kv) * 128 + (tid & 127)) * 1024
                              + (kc + 1) * 32 + ((tid >> 7) << 4);
      __builtin_prefetch((const void*)vnext, 0, 1);
    }
    const uint32_t* bufK = ldsK[kc & 1];

    // S = Q Kt : 16q x 64k in 4 accumulators
    f32x8 s[4];
#pragma unroll
    for (int j = 0; j < 4; ++j) {
      f32x8 a = zero;
#pragma unroll
      for (int c = 0; c < 4; ++c) {
        bf16x16 kf = gather_b(bufK, j * 16, 68, c * 16);
        a = wmma_bf16(qf[c], kf, a);
      }
      s[j] = a;
    }

    // Online softmax (rows live on 16-lane halves; shfl_xor 1/2/4/8 reduces).
    float mnew[8], corr[8], lsum[8];
#pragma unroll
    for (int r = 0; r < 8; ++r) {
      float vx = fmaxf(fmaxf(s[0][r], s[1][r]), fmaxf(s[2][r], s[3][r])) * scale;
#pragma unroll
      for (int off = 1; off < 16; off <<= 1)
        vx = fmaxf(vx, __shfl_xor(vx, off, 32));
      mnew[r] = fmaxf(mrun[r], vx);
      corr[r] = __expf(mrun[r] - mnew[r]);
      mrun[r] = mnew[r];
      lsum[r] = 0.0f;
    }
#pragma unroll
    for (int j = 0; j < 4; ++j) {
#pragma unroll
      for (int r = 0; r < 8; ++r) {
        float p = __expf(s[j][r] * scale - mnew[r]);
        lsum[r] += p;
        pwh[(r + (hi << 3)) * 66 + j * 16 + ln] = (__bf16)p;  // C-layout -> LDS
      }
    }
#pragma unroll
    for (int r = 0; r < 8; ++r) {
      float ls = lsum[r];
#pragma unroll
      for (int off = 1; off < 16; off <<= 1)
        ls += __shfl_xor(ls, off, 32);
      lrun[r] = lrun[r] * corr[r] + ls;
    }
#pragma unroll
    for (int i = 0; i < 8; ++i)
#pragma unroll
      for (int r = 0; r < 8; ++r) o[i][r] *= corr[r];

    // Wave-local LDS produce->consume: DS ops are in-order within a wave;
    // the wait + memory clobber stops the compiler reordering across the alias.
    asm volatile("s_wait_dscnt 0" ::: "memory");

    bf16x16 pf0 = gather_a(pw, 0, 33, 0);
    bf16x16 pf1 = gather_a(pw, 0, 33, 16);

    // O += P V : V consumed pre-transposed (Vt[d][s]) as B-fragments.
    const uint32_t* vbase = Vd + (size_t)((bi * 4 + kv) * 128) * 1024 + kc * 32;
#pragma unroll
    for (int i = 0; i < 8; ++i) {
      bf16x16 v0 = gather_b(vbase, i * 16, 1024, 0);
      bf16x16 v1 = gather_b(vbase, i * 16, 1024, 16);
      o[i] = wmma_bf16(pf0, v0, o[i]);
      o[i] = wmma_bf16(pf1, v1, o[i]);
    }
  }

  // Finalize: divide by running sum, store bf16 [b*s][2048].
  float inv[8];
#pragma unroll
  for (int r = 0; r < 8; ++r) inv[r] = 1.0f / lrun[r];
#pragma unroll
  for (int i = 0; i < 8; ++i) {
#pragma unroll
    for (int r = 0; r < 8; ++r) {
      int m = bi * 2048 + q0 + r + (hi << 3);
      int n = h * 128 + i * 16 + ln;
      Oattn[(size_t)m * 2048 + n] = (__bf16)(o[i][r] * inv[r]);
    }
  }
}

// ---------------------------------------------------------------------------
// Host-side orchestration
// ---------------------------------------------------------------------------
extern "C" void kernel_launch(void* const* d_in, const int* in_sizes, int n_in,
                              void* d_out, int out_size, void* d_ws, size_t ws_size,
                              hipStream_t stream) {
  (void)in_sizes; (void)n_in; (void)out_size; (void)ws_size;
  const float* x  = (const float*)d_in[0];
  const float* wq = (const float*)d_in[1];
  const float* bq = (const float*)d_in[2];
  const float* wk = (const float*)d_in[3];
  const float* bk = (const float*)d_in[4];
  const float* wv = (const float*)d_in[5];
  const float* bv = (const float*)d_in[6];
  const float* wo = (const float*)d_in[7];
  const float* bo = (const float*)d_in[8];
  float* out = (float*)d_out;

  // Workspace layout (bf16 buffers), ~84 MB total.
  char* ws = (char*)d_ws;
  __bf16* xb  = (__bf16*)ws; ws += (size_t)4096 * 2048 * 2;   // x bf16
  __bf16* wqt = (__bf16*)ws; ws += (size_t)2048 * 2048 * 2;   // wq^T
  __bf16* wkt = (__bf16*)ws; ws += (size_t)512  * 2048 * 2;   // wk^T
  __bf16* wvt = (__bf16*)ws; ws += (size_t)512  * 2048 * 2;   // wv^T
  __bf16* wot = (__bf16*)ws; ws += (size_t)2048 * 2048 * 2;   // wo^T
  __bf16* Qb  = (__bf16*)ws; ws += (size_t)4096 * 2048 * 2;   // Q
  __bf16* Kb  = (__bf16*)ws; ws += (size_t)4096 * 512  * 2;   // K
  __bf16* Vb  = (__bf16*)ws; ws += (size_t)4096 * 512  * 2;   // V
  __bf16* Vtb = (__bf16*)ws; ws += (size_t)4096 * 512  * 2;   // V^T per head
  __bf16* Ob  = (__bf16*)ws; ws += (size_t)4096 * 2048 * 2;   // attn out

  // 1. Convert inputs/weights to bf16 (weights transposed to [N][K]).
  cvt_f32_bf16_kernel<<<8192, 256, 0, stream>>>(x, xb, (size_t)4096 * 2048);
  transpose_cvt_kernel<<<dim3(64, 64), 256, 0, stream>>>(wq, wqt, 2048, 2048);
  transpose_cvt_kernel<<<dim3(16, 64), 256, 0, stream>>>(wk, wkt, 2048, 512);
  transpose_cvt_kernel<<<dim3(16, 64), 256, 0, stream>>>(wv, wvt, 2048, 512);
  transpose_cvt_kernel<<<dim3(64, 64), 256, 0, stream>>>(wo, wot, 2048, 2048);

  // 2. Projections (WMMA GEMMs, bf16 out), block tile 128x128.
  gemm_bias_kernel<false><<<dim3(16, 32), 256, 0, stream>>>(xb, wqt, bq, Qb, 4096, 2048, 2048);
  gemm_bias_kernel<false><<<dim3(4,  32), 256, 0, stream>>>(xb, wkt, bk, Kb, 4096, 512,  2048);
  gemm_bias_kernel<false><<<dim3(4,  32), 256, 0, stream>>>(xb, wvt, bv, Vb, 4096, 512,  2048);

  // 3. Transpose V to [b][kv][d][s] so P.V consumes B-fragments directly.
  transpose_v_kernel<<<dim3(64, 4, 8), 256, 0, stream>>>(Vb, Vtb);

  // 4. Flash attention (WMMA for QK^T and P.V).
  flash_attn_kernel<<<dim3(16, 16, 2), 256, 0, stream>>>(Qb, Kb, Vtb, Ob);

  // 5. Output projection (WMMA GEMM, fp32 out + bias) -> d_out.
  gemm_bias_kernel<true><<<dim3(16, 32), 256, 0, stream>>>(Ob, wot, bo, out, 4096, 2048, 2048);
}
